// WindowAttention_41188736368743
// MI455X (gfx1250) — matrix operand order repaired and checked by reading, not compiled, not run on hardware
//
#include <hip/hip_runtime.h>

typedef __attribute__((ext_vector_type(16))) __bf16 v16bf;
typedef __attribute__((ext_vector_type(8)))  __bf16 v8bf;
typedef __attribute__((ext_vector_type(4)))  __bf16 v4bf;
typedef __attribute__((ext_vector_type(8)))  float  v8f;

#define NTOK  49
#define NP    64
#define CDIM  128
#define HEADS 4
#define HD    32

// workspace layout (bytes)
#define WS_WQT   0          // bf16 [384][128]  = qkv_w^T
#define WS_PT    98304      // bf16 [128][128]  = proj_w^T
#define WS_BIAS  131072     // f32  [4][49][49] = gathered rel-pos bias
// LDS layout (bytes)
#define L_XB   0            // bf16 [64][128]  x staged (later reused as Y = concat-head out)
#define L_QB   16384        // bf16 [4][64][32]
#define L_KB   32768        // bf16 [4][64][32]
#define L_VT   49152        // bf16 [4][32][64] (v transposed: d-major)
#define L_LG   65536        // f32  [4][64][64] logits
#define L_AB   131072       // bf16 [4][64][64] softmax probs
#define SMEM_BYTES 163840

__device__ __forceinline__ v8f wmma_bf16(v16bf a, v16bf b, v8f c) {
  return __builtin_amdgcn_wmma_f32_16x16x32_bf16(false, a, false, b, (short)0, c,
                                                 false, false);
}

// A fragment: row-major [*, ld] bf16 matrix, tile origin (m0, k0), K=32.
// ISA 16-bit A 16x32 layout: elems 0..7 -> K = k0+half*8+i ; 8..15 -> +16.
__device__ __forceinline__ v16bf frag_A(const __bf16* M, int ld, int m0, int k0, int lane) {
  const int half = lane >> 4;
  const __bf16* p = M + (m0 + (lane & 15)) * ld + k0 + half * 8;
  v8bf lo = *(const v8bf*)(p);
  v8bf hi = *(const v8bf*)(p + 16);
  v16bf a;
#pragma unroll
  for (int i = 0; i < 8; ++i) { a[i] = lo[i]; a[i + 8] = hi[i]; }
  return a;
}

// B fragment, sourced from B^T stored row-major [N][K] (row = output col),
// tile origin (k0, n0). ISA B layout: elems 0..7 -> K = k0+half*8+i ; 8..15 -> +16.
__device__ __forceinline__ v16bf frag_B(const __bf16* BT, int ld, int k0, int n0, int lane) {
  const int half = lane >> 4;
  const __bf16* p = BT + (n0 + (lane & 15)) * ld + k0 + half * 8;
  v8bf lo = *(const v8bf*)(p);
  v8bf hi = *(const v8bf*)(p + 16);
  v16bf b;
#pragma unroll
  for (int i = 0; i < 8; ++i) { b[i] = lo[i]; b[i + 8] = hi[i]; }
  return b;
}

// ---- prep: weight transpose+bf16 convert, bias gather --------------------
__global__ __launch_bounds__(256) void swin_prep_kernel(
    const float* __restrict__ qkv_w, const float* __restrict__ proj_w,
    const float* __restrict__ bias_table, const int* __restrict__ rel_idx,
    __bf16* __restrict__ WqT, __bf16* __restrict__ PT, float* __restrict__ biasHNN) {
  const int i0 = blockIdx.x * blockDim.x + threadIdx.x;
  const int stride = gridDim.x * blockDim.x;
  for (int i = i0; i < 384 * 128; i += stride) {
    int n = i >> 7, k = i & 127;
    WqT[i] = (__bf16)qkv_w[k * 384 + n];
  }
  for (int i = i0; i < 128 * 128; i += stride) {
    int n = i >> 7, k = i & 127;
    PT[i] = (__bf16)proj_w[k * 128 + n];
  }
  for (int i = i0; i < HEADS * NTOK * NTOK; i += stride) {
    int h = i / (NTOK * NTOK), r = i % (NTOK * NTOK);
    biasHNN[i] = bias_table[rel_idx[r] * HEADS + h];
  }
}

// ---- fused window attention ----------------------------------------------
__global__ __launch_bounds__(256) void swin_attn_kernel(
    const float* __restrict__ x, const float* __restrict__ mask,
    const float* __restrict__ qkv_b, const float* __restrict__ proj_b,
    const __bf16* __restrict__ WqT, const __bf16* __restrict__ PT,
    const float* __restrict__ biasHNN, float* __restrict__ out) {
  extern __shared__ char smem[];
  __bf16* Xb = (__bf16*)(smem + L_XB);   // [64][128]; later reused as Y
  __bf16* Qb = (__bf16*)(smem + L_QB);   // [4][64][32]
  __bf16* Kb = (__bf16*)(smem + L_KB);   // [4][64][32]
  __bf16* Vt = (__bf16*)(smem + L_VT);   // [4][32][64] (v transposed: d-major)
  float*  Lg = (float*) (smem + L_LG);   // [4][64][64]
  __bf16* Ab = (__bf16*)(smem + L_AB);   // [4][64][64]

  const int b    = blockIdx.x;
  const int tid  = threadIdx.x;
  // force the per-wave index into an SGPR so tile math / q-k-v routing is scalar
  const int wave = __builtin_amdgcn_readfirstlane(tid >> 5);
  const int lane = tid & 31;
  const int half = lane >> 4;
  const int ln   = lane & 15;

  // 1) stage x -> bf16 with wide loads: 49*128 f32 = 1568 float4, then
  //    zero-pad rows 49..63 (element 6272 is 8-byte aligned in bf16).
  {
    const float4* x4 = (const float4*)(x + (size_t)b * NTOK * CDIM);
    for (int i = tid; i < (NTOK * CDIM) / 4; i += 256) {       // 1568
      float4 f = x4[i];
      v4bf o;
      o[0] = (__bf16)f.x; o[1] = (__bf16)f.y;
      o[2] = (__bf16)f.z; o[3] = (__bf16)f.w;
      *(v4bf*)(Xb + i * 4) = o;
    }
    unsigned long long* z = (unsigned long long*)(Xb + NTOK * CDIM);
    for (int i = tid; i < (NP - NTOK) * CDIM / 4; i += 256)    // 480
      z[i] = 0ull;
  }
  __syncthreads();

  // 2) QKV GEMM: [64x128] @ [128x384] ; 4x24 tiles, K = 4x32
  for (int t = wave; t < 96; t += 8) {
    const int mt = t / 24, nt = t % 24;          // scalar
    v8f c = {};
#pragma unroll
    for (int kt = 0; kt < 4; ++kt) {
      v16bf a  = frag_A(Xb, CDIM, mt * 16, kt * 32, lane);
      v16bf bb = frag_B(WqT, CDIM, kt * 32, nt * 16, lane);
      c = wmma_bf16(a, bb, c);
    }
    const float bias = qkv_b[nt * 16 + ln];
    const int j = nt >> 3;                       // scalar: 0=q 1=k 2=v
    const int h = (nt >> 1) & 3;                 // scalar head
    const int d = ((nt & 1) << 4) + ln;          // 0..31
    if (j == 0) {
#pragma unroll
      for (int r = 0; r < 8; ++r) {
        int m = mt * 16 + half * 8 + r;
        Qb[(h * NP + m) * HD + d] = (__bf16)(c[r] + bias);
      }
    } else if (j == 1) {
#pragma unroll
      for (int r = 0; r < 8; ++r) {
        int m = mt * 16 + half * 8 + r;
        Kb[(h * NP + m) * HD + d] = (__bf16)(c[r] + bias);
      }
    } else {
#pragma unroll
      for (int r = 0; r < 8; ++r) {
        int m = mt * 16 + half * 8 + r;
        Vt[(h * HD + d) * NP + m] = (__bf16)(c[r] + bias);
      }
    }
  }
  __syncthreads();

  // 3) logits = scale * Q K^T + bias + mask ; per head 4x4 tiles, K=32
  const float scale = 0.17677669529663687f;   // 32^-0.5
  const float* mw = mask + (size_t)(b & 63) * NTOK * NTOK;
  for (int t = wave; t < 64; t += 8) {
    const int h = t >> 4, mt = (t >> 2) & 3, nt = t & 3;   // scalar
    v16bf a  = frag_A(Qb + h * NP * HD, HD, mt * 16, 0, lane);
    v16bf bb = frag_B(Kb + h * NP * HD, HD, 0, nt * 16, lane);
    v8f c = {};
    c = wmma_bf16(a, bb, c);
    const int m = nt * 16 + ln;
#pragma unroll
    for (int r = 0; r < 8; ++r) {
      int n = mt * 16 + half * 8 + r;
      float v;
      if (m >= NTOK) {
        v = -1e30f;                           // mask padded key columns
      } else {
        v = c[r] * scale;
        if (n < NTOK)
          v += biasHNN[(h * NTOK + n) * NTOK + m] + mw[n * NTOK + m];
      }
      Lg[(h * NP + n) * NP + m] = v;
    }
  }
  __syncthreads();

  // 4) softmax over keys; one row per thread, rotated column order
  {
    const int h = tid >> 6, n = tid & 63;
    float* row = Lg + (h * NP + n) * NP;
    float mx = -1e30f;
    for (int i = 0; i < NP; ++i) { int m = (tid + i) & 63; mx = fmaxf(mx, row[m]); }
    float s = 0.f;
    for (int i = 0; i < NP; ++i) {
      int m = (tid + i) & 63;
      float e = __expf(row[m] - mx);
      row[m] = e; s += e;
    }
    const float inv = 1.f / s;
    __bf16* arow = Ab + (h * NP + n) * NP;
    for (int i = 0; i < NP; ++i) { int m = (tid + i) & 63; arow[m] = (__bf16)(row[m] * inv); }
  }
  __syncthreads();

  // 5) out_h = P @ V : per head 4x2 tiles, K = 2x32 ; write concat-heads Y
  for (int t = wave; t < 32; t += 8) {
    const int h = t >> 3, mt = (t >> 1) & 3, nt = t & 1;   // scalar
    v8f c = {};
#pragma unroll
    for (int kt = 0; kt < 2; ++kt) {
      v16bf a  = frag_A(Ab + h * NP * NP, NP, mt * 16, kt * 32, lane);
      v16bf bb = frag_B(Vt + h * HD * NP, NP, kt * 32, nt * 16, lane);
      c = wmma_bf16(a, bb, c);
    }
    const int d = nt * 16 + ln;
#pragma unroll
    for (int r = 0; r < 8; ++r) {
      int n = mt * 16 + half * 8 + r;
      Xb[n * CDIM + h * HD + d] = (__bf16)c[r];   // Y[n][h*32+d]
    }
  }
  __syncthreads();

  // 6) O = Y @ proj_w + proj_b ; 4x8 tiles, K = 4x32 ; store rows < 49
  float* ob = out + (size_t)b * NTOK * CDIM;
  for (int t = wave; t < 32; t += 8) {
    const int mt = t >> 3, nt = t & 7;           // scalar
    v8f c = {};
#pragma unroll
    for (int kt = 0; kt < 4; ++kt) {
      v16bf a  = frag_A(Xb, CDIM, mt * 16, kt * 32, lane);
      v16bf bb = frag_B(PT, CDIM, kt * 32, nt * 16, lane);
      c = wmma_bf16(a, bb, c);
    }
    const int col = nt * 16 + ln;
    const float pb = proj_b[col];
#pragma unroll
    for (int r = 0; r < 8; ++r) {
      int n = mt * 16 + half * 8 + r;
      if (n < NTOK) ob[n * CDIM + col] = c[r] + pb;
    }
  }
}

extern "C" void kernel_launch(void* const* d_in, const int* in_sizes, int n_in,
                              void* d_out, int out_size, void* d_ws, size_t ws_size,
                              hipStream_t stream) {
  (void)in_sizes; (void)n_in; (void)out_size; (void)ws_size;
  const float* x          = (const float*)d_in[0];
  const float* mask       = (const float*)d_in[1];
  const float* qkv_w      = (const float*)d_in[2];
  const float* qkv_b      = (const float*)d_in[3];
  const float* proj_w     = (const float*)d_in[4];
  const float* proj_b     = (const float*)d_in[5];
  const float* bias_table = (const float*)d_in[6];
  const int*   rel_idx    = (const int*)d_in[7];

  char* ws = (char*)d_ws;
  __bf16* WqT    = (__bf16*)(ws + WS_WQT);
  __bf16* PT     = (__bf16*)(ws + WS_PT);
  float*  biasHNN = (float*)(ws + WS_BIAS);

  (void)hipFuncSetAttribute((const void*)swin_attn_kernel,
                            hipFuncAttributeMaxDynamicSharedMemorySize, SMEM_BYTES);

  swin_prep_kernel<<<64, 256, 0, stream>>>(qkv_w, proj_w, bias_table, rel_idx,
                                           WqT, PT, biasHNN);
  swin_attn_kernel<<<4096, 256, SMEM_BYTES, stream>>>(x, mask, qkv_b, proj_b,
                                                      WqT, PT, biasHNN, (float*)d_out);
}
